// new_LSTM_30365418783418
// MI455X (gfx1250) — matrix operand order repaired and checked by reading, not compiled, not run on hardware
//
#include <hip/hip_runtime.h>
#include <hip/hip_bf16.h>
#include <math.h>

// ---------------- problem dims ----------------
#define BB 128
#define TT 256
#define II 512
#define HH 1024
#define BT (BB*TT)   // 32768 rows for time-parallel GEMMs

typedef __attribute__((ext_vector_type(16))) __bf16 v16bf;
typedef __attribute__((ext_vector_type(8)))  float  v8f;

__device__ __forceinline__ float sigf_(float x) { return 1.0f / (1.0f + __expf(-x)); }

// D = A(16x32 bf16) x B(32x16 bf16) + C, f32 accumulate
__device__ __forceinline__ v8f wmma_bf16(v16bf a, v16bf b, v8f c) {
    return __builtin_amdgcn_wmma_f32_16x16x32_bf16(
        /*neg_a=*/false, a, /*neg_b=*/false, b,
        /*c_mod=*/(short)0, c, /*reuse_a=*/false, /*reuse_b=*/false);
}

// ---- CDNA5 async copy: 16B global -> LDS, tracked by ASYNCcnt ----
// Generic pointers derived from LDS have their LDS byte offset in addr[31:0].
__device__ __forceinline__ void async_ld16(void* lds, const void* g) {
    asm volatile("global_load_async_to_lds_b128 %0, %1, off"
                 :: "v"((unsigned int)(size_t)lds), "v"((unsigned long long)(size_t)g)
                 : "memory");
}
__device__ __forceinline__ void wait_async0() {
    asm volatile("s_wait_asynccnt 0x0" ::: "memory");
}

// A-fragment (16 rows x 32 K, bf16) per ISA wave32 layout, from a padded buffer.
// lanes 0-15: K k0..k0+7 (dw0-3), k0+16..k0+23 (dw4-7); lanes 16-31: +8.
__device__ __forceinline__ v16bf load_a_any(const __bf16* base, int stride,
                                            int k0, int lane) {
    int row = lane & 15;
    int kb  = (lane & 16) ? 8 : 0;
    const __bf16* p = base + (size_t)row * stride + k0 + kb;
    union { v16bf v; float4 f[2]; } u;
    u.f[0] = *(const float4*)(p);
    u.f[1] = *(const float4*)(p + 16);
    return u.v;
}

// B-fragment from row-major (N,K) matrix: col n = n0+(lane&15);
// lanes 0-15 hold K k0..k0+7, k0+16..k0+23; lanes 16-31: +8 (matches A).
__device__ __forceinline__ v16bf load_b_any(const __bf16* base, int stride,
                                            int n0, int k0, int lane) {
    int col = lane & 15;
    int kb  = (lane & 16) ? 8 : 0;
    const __bf16* p = base + (size_t)(n0 + col) * stride + k0 + kb;
    union { v16bf v; float4 f[2]; } u;
    u.f[0] = *(const float4*)(p);
    u.f[1] = *(const float4*)(p + 16);
    return u.v;
}

// ---------------- utility kernels ----------------
__global__ void k_cvt_bf16(const float* __restrict__ s, __bf16* __restrict__ d, int n) {
    int i = blockIdx.x * blockDim.x + threadIdx.x;
    if (i < n) d[i] = (__bf16)s[i];
}

__global__ void k_copy_f32(const float* __restrict__ s, float* __restrict__ d, int n) {
    int i = blockIdx.x * blockDim.x + threadIdx.x;
    if (i < n) d[i] = s[i];
}

__global__ void k_finalize(const float* __restrict__ hlast, const float* __restrict__ c,
                           float* __restrict__ out_h, float* __restrict__ out_c, int n) {
    int i = blockIdx.x * blockDim.x + threadIdx.x;
    if (i < n) { out_h[i] = hlast[i]; out_c[i] = c[i]; }
}

// epilogue helper: write one 16x16 f32 tile (per-lane 8 rows x 1 col)
__device__ __forceinline__ void store_tile(v8f acc, float* __restrict__ C,
                                           const float* __restrict__ bias,
                                           int m0, int n, int N, int rb,
                                           int addC, int act) {
    float bv = bias ? bias[n] : 0.0f;
#pragma unroll
    for (int r = 0; r < 8; ++r) {
        size_t idx = (size_t)(m0 + rb + r) * N + n;
        float v = acc[r] + bv;
        if (addC) v += C[idx];
        if (act)  v = tanhf(v);
        C[idx] = v;
    }
}

// ---------------- WMMA GEMM: C = act(A @ W^T + bias [+ Cin]) ----------------
// A (M,K) bf16 row-major, W (N,K) bf16 row-major, C (M,N) f32.
// Block 256 thr = 8 waves; block tile M=128 x N=128 x K=32; LDS double-buffered
// via async-to-LDS. Wave grid 4(M) x 2(N); each wave computes a 32x64 strip:
// 2 A-frags x 4 B-frags -> 8 independent accumulator chains (8 WMMA per 12
// ds_load_b128; 64 FLOP per global byte at block level).
#define GBM 128
#define GBN 128
#define GBK 32
#define GPAD 40   // padded LDS row stride (elems) -> conflict-free b128 reads

__global__ __launch_bounds__(256) void k_wmma_gemm(
    const __bf16* __restrict__ A, const __bf16* __restrict__ W,
    const float* __restrict__ bias, float* __restrict__ C,
    int M, int N, int K, int addC, int act)
{
    __shared__ __bf16 As[2][GBM * GPAD];
    __shared__ __bf16 Bs[2][GBN * GPAD];

    int tid  = threadIdx.x;
    int wave = tid >> 5, lane = tid & 31;
    int wm = wave >> 1;          // 0..3 -> 32-row strip
    int wn = wave & 1;           // 0..1 -> 64-col strip
    int mblk = blockIdx.y * GBM;
    int nblk = blockIdx.x * GBN;

    // staging: A tile 128x32 = 512 x16B chunks (2/thread); B tile same.
    int srow = tid >> 2;                 // 0..63
    int scol = (tid & 3) * 8;            // elem offset 0/8/16/24
    const __bf16* gA0 = A + (size_t)(mblk + srow) * K + scol;
    const __bf16* gA1 = A + (size_t)(mblk + srow + 64) * K + scol;
    const __bf16* gB0 = W + (size_t)(nblk + srow) * K + scol;
    const __bf16* gB1 = W + (size_t)(nblk + srow + 64) * K + scol;

    v8f a00 = {}, a01 = {}, a02 = {}, a03 = {};
    v8f a10 = {}, a11 = {}, a12 = {}, a13 = {};
    int nk = K / GBK;

    // prime buffer 0
    async_ld16(&As[0][srow * GPAD + scol],        gA0);
    async_ld16(&As[0][(srow + 64) * GPAD + scol], gA1);
    async_ld16(&Bs[0][srow * GPAD + scol],        gB0);
    async_ld16(&Bs[0][(srow + 64) * GPAD + scol], gB1);

    for (int ki = 0; ki < nk; ++ki) {
        wait_async0();           // my copies for chunk ki have landed
        __syncthreads();         // everyone's copies visible; prev compute done
        if (ki + 1 < nk) {
            int nb = (ki + 1) & 1;
            int k0 = (ki + 1) * GBK;
            async_ld16(&As[nb][srow * GPAD + scol],        gA0 + k0);
            async_ld16(&As[nb][(srow + 64) * GPAD + scol], gA1 + k0);
            async_ld16(&Bs[nb][srow * GPAD + scol],        gB0 + k0);
            async_ld16(&Bs[nb][(srow + 64) * GPAD + scol], gB1 + k0);
        }
        int buf = ki & 1;
        v16bf fa0 = load_a_any(&As[buf][(size_t)(wm * 32 +  0) * GPAD], GPAD, 0, lane);
        v16bf fa1 = load_a_any(&As[buf][(size_t)(wm * 32 + 16) * GPAD], GPAD, 0, lane);
        v16bf fb0 = load_b_any(&Bs[buf][0], GPAD, wn * 64 +  0, 0, lane);
        v16bf fb1 = load_b_any(&Bs[buf][0], GPAD, wn * 64 + 16, 0, lane);
        v16bf fb2 = load_b_any(&Bs[buf][0], GPAD, wn * 64 + 32, 0, lane);
        v16bf fb3 = load_b_any(&Bs[buf][0], GPAD, wn * 64 + 48, 0, lane);
        a00 = wmma_bf16(fa0, fb0, a00);
        a01 = wmma_bf16(fa0, fb1, a01);
        a02 = wmma_bf16(fa0, fb2, a02);
        a03 = wmma_bf16(fa0, fb3, a03);
        a10 = wmma_bf16(fa1, fb0, a10);
        a11 = wmma_bf16(fa1, fb1, a11);
        a12 = wmma_bf16(fa1, fb2, a12);
        a13 = wmma_bf16(fa1, fb3, a13);
    }

    int col = lane & 15;
    int rb  = (lane & 16) ? 8 : 0;
    int m0  = mblk + wm * 32;
    int nw  = nblk + wn * 64 + col;
    store_tile(a00, C, bias, m0,      nw +  0, N, rb, addC, act);
    store_tile(a01, C, bias, m0,      nw + 16, N, rb, addC, act);
    store_tile(a02, C, bias, m0,      nw + 32, N, rb, addC, act);
    store_tile(a03, C, bias, m0,      nw + 48, N, rb, addC, act);
    store_tile(a10, C, bias, m0 + 16, nw +  0, N, rb, addC, act);
    store_tile(a11, C, bias, m0 + 16, nw + 16, N, rb, addC, act);
    store_tile(a12, C, bias, m0 + 16, nw + 32, N, rb, addC, act);
    store_tile(a13, C, bias, m0 + 16, nw + 48, N, rb, addC, act);
}

// ---------------- one recurrent timestep ----------------
// Block 256 thr (8 waves): tile M=16 (batch) x N=128; grid (H/128, B/16).
// h tile (16x1024 bf16 = 32KB) is async-staged into LDS once and shared by all
// waves; 5 gate GEMMs give 5 independent WMMA chains per wave; pointwise LSTM
// update fully in registers.
#define HPAD 1032  // padded LDS row stride for h tile

__global__ __launch_bounds__(256) void k_lstm_step(
    const __bf16* __restrict__ h_in, __bf16* __restrict__ h_out,
    float* __restrict__ c, float* __restrict__ h_last,
    const float* __restrict__ Pi, const float* __restrict__ Pf,
    const float* __restrict__ Pg, const float* __restrict__ Po,
    const float* __restrict__ Pd,
    const __bf16* __restrict__ Whi, const __bf16* __restrict__ Whf,
    const __bf16* __restrict__ Whg, const __bf16* __restrict__ Who,
    const __bf16* __restrict__ Whd,
    __bf16* __restrict__ hs_bf, int t)
{
    __shared__ __bf16 Hs[16 * HPAD];

    int tid  = threadIdx.x;
    int wave = tid >> 5, lane = tid & 31;
    int n0 = blockIdx.x * 128 + wave * 16;
    int m0 = blockIdx.y * 16;

    // stage 16x1024 h tile: 2048 x16B chunks, 8 per thread (async-to-LDS)
    for (int j = tid; j < 2048; j += 256) {
        int row  = j >> 7;
        int coff = (j & 127) * 8;
        async_ld16(&Hs[row * HPAD + coff],
                   h_in + (size_t)(m0 + row) * HH + coff);
    }
    wait_async0();
    __syncthreads();

    v8f ai = {}, af = {}, ag = {}, ao = {}, ad = {};
    for (int k0 = 0; k0 < HH; k0 += 32) {
        v16bf a = load_a_any(Hs, HPAD, k0, lane);
        ai = wmma_bf16(a, load_b_any(Whi, HH, n0, k0, lane), ai);
        af = wmma_bf16(a, load_b_any(Whf, HH, n0, k0, lane), af);
        ag = wmma_bf16(a, load_b_any(Whg, HH, n0, k0, lane), ag);
        ao = wmma_bf16(a, load_b_any(Who, HH, n0, k0, lane), ao);
        ad = wmma_bf16(a, load_b_any(Whd, HH, n0, k0, lane), ad);
    }

    int col = lane & 15;
    int rb  = (lane & 16) ? 8 : 0;
    int n   = n0 + col;
#pragma unroll
    for (int r = 0; r < 8; ++r) {
        int b = m0 + rb + r;                         // batch index
        size_t off = ((size_t)b * TT + t) * HH + n;  // (b,t,h) for P / hs
        size_t bh  = (size_t)b * HH + n;             // (b,h) for state
        float i_t = sigf_(Pi[off] + ai[r]);
        float f_t = sigf_(Pf[off] + af[r]);
        float g_t = tanhf(Pg[off] + ag[r]);
        float o_t = sigf_(Po[off] + ao[r]);
        float d_t = sigf_(Pd[off] + ad[r]);
        float cn  = f_t * c[bh] + i_t * g_t + d_t;
        float hn  = o_t * tanhf(cn);
        c[bh]      = cn;
        h_out[bh]  = (__bf16)hn;
        hs_bf[off] = (__bf16)hn;
        if (t == TT - 1) h_last[bh] = hn;
    }
}

// ---------------- host orchestration ----------------
extern "C" void kernel_launch(void* const* d_in, const int* in_sizes, int n_in,
                              void* d_out, int out_size, void* d_ws, size_t ws_size,
                              hipStream_t stream)
{
    const float* x      = (const float*)d_in[0];
    const float* x_d    = (const float*)d_in[1];
    const float* hidden = (const float*)d_in[2];
    const float* cell   = (const float*)d_in[3];
    const float* Wii = (const float*)d_in[4];
    const float* Wif = (const float*)d_in[5];
    const float* Wig = (const float*)d_in[6];
    const float* Wio = (const float*)d_in[7];
    const float* Wid = (const float*)d_in[8];
    const float* Wro = (const float*)d_in[9];
    const float* Whi = (const float*)d_in[10];
    const float* Whf = (const float*)d_in[11];
    const float* Whg = (const float*)d_in[12];
    const float* Who = (const float*)d_in[13];
    const float* Whd = (const float*)d_in[14];
    const float* bi  = (const float*)d_in[15];
    const float* bf  = (const float*)d_in[16];
    const float* bg  = (const float*)d_in[17];
    const float* bo  = (const float*)d_in[18];
    const float* bd  = (const float*)d_in[19];
    const float* Wrb = (const float*)d_in[20];

    // workspace carve-up (256B aligned)
    char*  ws  = (char*)d_ws;
    size_t off = 0;
    auto alloc = [&](size_t bytes) -> char* {
        char* p = ws + off;
        off += (bytes + 255) & ~(size_t)255;
        return p;
    };
    __bf16* xb   = (__bf16*)alloc((size_t)BT * II * 2);
    __bf16* xdb  = (__bf16*)alloc((size_t)BT * II * 2);
    __bf16* wiib = (__bf16*)alloc((size_t)HH * II * 2);
    __bf16* wifb = (__bf16*)alloc((size_t)HH * II * 2);
    __bf16* wigb = (__bf16*)alloc((size_t)HH * II * 2);
    __bf16* wiob = (__bf16*)alloc((size_t)HH * II * 2);
    __bf16* widb = (__bf16*)alloc((size_t)HH * II * 2);
    __bf16* wrob = (__bf16*)alloc((size_t)HH * II * 2);
    __bf16* whib = (__bf16*)alloc((size_t)HH * HH * 2);
    __bf16* whfb = (__bf16*)alloc((size_t)HH * HH * 2);
    __bf16* whgb = (__bf16*)alloc((size_t)HH * HH * 2);
    __bf16* whob = (__bf16*)alloc((size_t)HH * HH * 2);
    __bf16* whdb = (__bf16*)alloc((size_t)HH * HH * 2);
    float*  P0   = (float*) alloc((size_t)BT * HH * 4);
    float*  P1   = (float*) alloc((size_t)BT * HH * 4);
    float*  P2   = (float*) alloc((size_t)BT * HH * 4);
    float*  P3   = (float*) alloc((size_t)BT * HH * 4);
    float*  P4   = (float*) alloc((size_t)BT * HH * 4);
    __bf16* hsb  = (__bf16*)alloc((size_t)BT * HH * 2);
    float*  cbuf = (float*) alloc((size_t)BB * HH * 4);
    float*  hlst = (float*) alloc((size_t)BB * HH * 4);
    __bf16* hb0  = (__bf16*)alloc((size_t)BB * HH * 2);
    __bf16* hb1  = (__bf16*)alloc((size_t)BB * HH * 2);

    auto cvt = [&](const float* s, __bf16* d, int n) {
        k_cvt_bf16<<<(n + 255) / 256, 256, 0, stream>>>(s, d, n);
    };
    cvt(x,   xb,  BT * II);
    cvt(x_d, xdb, BT * II);
    cvt(Wii, wiib, HH * II);  cvt(Wif, wifb, HH * II);
    cvt(Wig, wigb, HH * II);  cvt(Wio, wiob, HH * II);
    cvt(Wid, widb, HH * II);  cvt(Wro, wrob, HH * II);
    cvt(Whi, whib, HH * HH);  cvt(Whf, whfb, HH * HH);
    cvt(Whg, whgb, HH * HH);  cvt(Who, whob, HH * HH);
    cvt(Whd, whdb, HH * HH);
    cvt(hidden, hb0, BB * HH);
    k_copy_f32<<<(BB * HH + 255) / 256, 256, 0, stream>>>(cell, cbuf, BB * HH);

    // time-parallel input projections: P = x @ W^T + b
    dim3 gIn(HH / GBN, BT / GBM);
    k_wmma_gemm<<<gIn, 256, 0, stream>>>(xb,  wiib, bi, P0, BT, HH, II, 0, 0);
    k_wmma_gemm<<<gIn, 256, 0, stream>>>(xb,  wifb, bf, P1, BT, HH, II, 0, 0);
    k_wmma_gemm<<<gIn, 256, 0, stream>>>(xb,  wigb, bg, P2, BT, HH, II, 0, 0);
    k_wmma_gemm<<<gIn, 256, 0, stream>>>(xb,  wiob, bo, P3, BT, HH, II, 0, 0);
    k_wmma_gemm<<<gIn, 256, 0, stream>>>(xdb, widb, bd, P4, BT, HH, II, 0, 0);

    // sequential recurrence: one launch per timestep (stream order = dependency)
    dim3 gStep(HH / 128, BB / 16);
    for (int t = 0; t < TT; ++t) {
        const __bf16* hin  = (t & 1) ? hb1 : hb0;
        __bf16*       hout = (t & 1) ? hb0 : hb1;
        k_lstm_step<<<gStep, 256, 0, stream>>>(
            hin, hout, cbuf, hlst, P0, P1, P2, P3, P4,
            whib, whfb, whgb, whob, whdb, hsb, t);
    }

    // readout: y = tanh(x @ Wro^T + Wrb + hs @ Who^T), written into d_out
    float* y = (float*)d_out;
    k_wmma_gemm<<<gIn, 256, 0, stream>>>(xb,  wrob, Wrb,     y, BT, HH, II, 0, 0);
    k_wmma_gemm<<<gIn, 256, 0, stream>>>(hsb, whob, nullptr, y, BT, HH, HH, 1, 1);

    // h_last, c_last appended after y
    float* out_h = y + (size_t)BT * HH;
    float* out_c = out_h + (size_t)BB * HH;
    k_finalize<<<(BB * HH + 255) / 256, 256, 0, stream>>>(hlst, cbuf, out_h, out_c, BB * HH);
}
